// ARDecoder_78975858639031
// MI455X (gfx1250) — matrix operand order repaired and checked by reading, not compiled, not run on hardware
//
#include <hip/hip_runtime.h>
#include <hip/hip_bf16.h>
#include <stdint.h>

// ---------------- model constants ----------------
constexpr int kD      = 512;
constexpr int kNH     = 16;
constexpr int kHD     = 32;
constexpr int kDFF    = 2048;
constexpr int kNL     = 6;
constexpr int kSEQ    = 26;           // MAXLEN + 1
constexpr int kMAXLEN = 25;
constexpr int kBS     = 32;
constexpr int kVFULL  = 6625;
constexpr int kVOUT   = 6623;         // V - 2
constexpr int kVPAD   = 6656;         // padded to multiple of 64 for WMMA tiles
constexpr int kBOS    = kVFULL - 2;
constexpr int kIGN    = kVFULL - 1;
constexpr int kM      = kBS * kSEQ;   // 832 rows (multiple of 16)
constexpr float kEmbScale = 22.62741699796952f;   // sqrt(512)
constexpr float kAttScale = 0.17677669529663687f; // 1/sqrt(32)

typedef __attribute__((ext_vector_type(16))) _Float16 v16h;
typedef __attribute__((ext_vector_type(8)))  _Float16 v8h;
typedef __attribute__((ext_vector_type(8)))  float    v8f;

union V16 { v16h v; v8h h[2]; };

// ---------------- fp32 -> fp16 conversions ----------------
__global__ void cvt_kernel(const float* __restrict__ s, _Float16* __restrict__ d, size_t n) {
    for (size_t i = (size_t)blockIdx.x * blockDim.x + threadIdx.x; i < n;
         i += (size_t)gridDim.x * blockDim.x)
        d[i] = (_Float16)s[i];
}

// pads rows [validRows, totalRows) with zeros (vocab projection padding)
__global__ void cvt_pad_kernel(const float* __restrict__ s, _Float16* __restrict__ d,
                               int validRows, int totalRows, int k) {
    size_t n = (size_t)totalRows * k;
    for (size_t i = (size_t)blockIdx.x * blockDim.x + threadIdx.x; i < n;
         i += (size_t)gridDim.x * blockDim.x) {
        int row = (int)(i / (size_t)k);
        d[i] = (row < validRows) ? (_Float16)s[i] : (_Float16)0.0f;
    }
}

__global__ void init_dec_kernel(int* __restrict__ dec) {
    int idx = blockIdx.x * blockDim.x + threadIdx.x;
    if (idx < kBS * kSEQ) dec[idx] = (idx % kSEQ == 0) ? kBOS : kIGN;
}

// ---------------- embedding: x = emb[dec]*sqrt(D) + src + pos ----------------
__global__ __launch_bounds__(128) void embed_kernel(
    const float* __restrict__ src, const float* __restrict__ pos,
    const float* __restrict__ emb, const int* __restrict__ dec,
    float* __restrict__ xf, _Float16* __restrict__ xh)
{
    int row = blockIdx.x;            // b*SEQ + n
    int n   = row % kSEQ;
    int tok = dec[row];
    const float* er = emb + (size_t)tok * kD;
    const float* sr = src + (size_t)row * kD;
    const float* pr = pos + (size_t)n * kD;
    for (int d = threadIdx.x; d < kD; d += 128) {
        float v = er[d] * kEmbScale + sr[d] + pr[d];
        xf[(size_t)row * kD + d] = v;
        xh[(size_t)row * kD + d] = (_Float16)v;
    }
}

// ---------------- WMMA GEMM: C[M,N] = A[M,K](f16) @ W[N,K]^T(f16) + bias ----------------
// One wave computes a 16x64 tile (4 accumulators sharing the A fragment).
// A fragment (16x32 f16): lane half sel=lane>>4 holds 16 contiguous K at sel*16.
// B fragment (32x16 f16): lane holds 16 contiguous K at sel*16 for column n = lane&15,
// which is exactly row (n) of the row-major [N,K] weight -> x @ W.T needs no transpose.
__global__ __launch_bounds__(128) void gemm_wmma_kernel(
    const _Float16* __restrict__ A, int lda,
    const _Float16* __restrict__ W,          // [N, K] row-major
    const float* __restrict__ bias,          // nullable
    float* __restrict__ Cf,                  // nullable fp32 out [M,N]
    _Float16* __restrict__ Ch,               // nullable fp16 out [M,N]
    int M, int N, int K, int flags)          // flags bit0: relu
{
    const int lane = threadIdx.x;            // 0..31
    const int wave = threadIdx.y;            // 0..3
    const int m0 = blockIdx.x * 16;
    const int n0 = (blockIdx.y * 4 + wave) * 64;
    if (n0 >= N) return;                     // wave-uniform; EXEC stays all-ones
    const int sel = lane >> 4;
    const int l16 = lane & 15;

    const _Float16* arow = A + (size_t)(m0 + l16) * lda;
    const _Float16* wb   = W + (size_t)(n0 + l16) * K;
    const size_t wstride16 = (size_t)16 * K;

    v8f c0 = {}, c1 = {}, c2 = {}, c3 = {};
    for (int k0 = 0; k0 < K; k0 += 32) {
        V16 a;
        a.h[0] = *(const v8h*)(arow + k0 + sel * 16);
        a.h[1] = *(const v8h*)(arow + k0 + sel * 16 + 8);
        const _Float16* wp = wb + k0 + sel * 16;
        v16h b0 = *(const v16h*)(wp);
        v16h b1 = *(const v16h*)(wp + wstride16);
        v16h b2 = *(const v16h*)(wp + 2 * wstride16);
        v16h b3 = *(const v16h*)(wp + 3 * wstride16);
        c0 = __builtin_amdgcn_wmma_f32_16x16x32_f16(false, a.v, false, b0, (short)0, c0, false, false);
        c1 = __builtin_amdgcn_wmma_f32_16x16x32_f16(false, a.v, false, b1, (short)0, c1, false, false);
        c2 = __builtin_amdgcn_wmma_f32_16x16x32_f16(false, a.v, false, b2, (short)0, c2, false, false);
        c3 = __builtin_amdgcn_wmma_f32_16x16x32_f16(false, a.v, false, b3, (short)0, c3, false, false);
    }

    const bool relu = (flags & 1) != 0;
    #pragma unroll
    for (int r = 0; r < 8; ++r) {
        const int m = m0 + r + 8 * sel;      // C layout: VGPR r -> M = r (lanes 0-15), r+8 (16-31)
        const size_t rowoff = (size_t)m * N;
        float vr[4] = { c0[r], c1[r], c2[r], c3[r] };
        #pragma unroll
        for (int nt = 0; nt < 4; ++nt) {
            const int n = n0 + nt * 16 + l16;
            float v = vr[nt];
            if (bias) v += bias[n];
            if (relu) v = fmaxf(v, 0.0f);
            if (Cf) Cf[rowoff + n] = v;
            if (Ch) Ch[rowoff + n] = (_Float16)v;
        }
    }
}

// ---------------- causal attention, one block per (b, h); tiny -> VALU+LDS ----------------
__global__ __launch_bounds__(32) void attn_kernel(const float* __restrict__ qkv, // [B,SEQ,3*D]
                                                  _Float16* __restrict__ oh)     // [B,SEQ,D] f16
{
    __shared__ float Ks[kSEQ][kHD];
    __shared__ float Vs[kSEQ][kHD];
    const int b = blockIdx.x >> 4;   // /NH
    const int h = blockIdx.x & 15;   // %NH
    const float* base = qkv + (size_t)b * kSEQ * (3 * kD);
    for (int i = threadIdx.x; i < kSEQ * kHD; i += 32) {
        int s = i / kHD, d = i % kHD;
        Ks[s][d] = base[(size_t)s * (3 * kD) + kD     + h * kHD + d];
        Vs[s][d] = base[(size_t)s * (3 * kD) + 2 * kD + h * kHD + d];
    }
    __syncthreads();
    const int qi = threadIdx.x;
    if (qi >= kSEQ) return;

    const float* qp = base + (size_t)qi * (3 * kD) + h * kHD;
    float qv[kHD];
    #pragma unroll
    for (int d = 0; d < kHD; ++d) qv[d] = qp[d];

    float sc[kSEQ];
    float mx = -3.4e38f;
    #pragma unroll
    for (int j = 0; j < kSEQ; ++j) {
        float s = 0.0f;
        #pragma unroll
        for (int d = 0; d < kHD; ++d) s += qv[d] * Ks[j][d];
        s = (j <= qi) ? s * kAttScale : -1e30f;   // causal mask
        sc[j] = s;
        mx = fmaxf(mx, s);
    }
    float sum = 0.0f;
    #pragma unroll
    for (int j = 0; j < kSEQ; ++j) { sc[j] = __expf(sc[j] - mx); sum += sc[j]; }
    const float inv = 1.0f / sum;

    float acc[kHD];
    #pragma unroll
    for (int d = 0; d < kHD; ++d) acc[d] = 0.0f;
    #pragma unroll
    for (int j = 0; j < kSEQ; ++j) {
        const float w = sc[j] * inv;
        #pragma unroll
        for (int d = 0; d < kHD; ++d) acc[d] += w * Vs[j][d];
    }
    _Float16* op = oh + ((size_t)b * kSEQ + qi) * kD + h * kHD;
    #pragma unroll
    for (int d = 0; d < kHD; ++d) op[d] = (_Float16)acc[d];
}

// ---------------- x = LN(x + y); writes f32 + f16 copies; one wave per row ----------------
__global__ __launch_bounds__(32) void add_ln_kernel(
    float* __restrict__ x, const float* __restrict__ y,
    const float* __restrict__ g, const float* __restrict__ bt,
    _Float16* __restrict__ xh)
{
    const int row = blockIdx.x;
    float* xr = x + (size_t)row * kD;
    const float* yr = y + (size_t)row * kD;
    float vals[16];
    float s = 0.0f;
    #pragma unroll
    for (int i = 0; i < 16; ++i) {
        int d = threadIdx.x + i * 32;
        float v = xr[d] + yr[d];
        vals[i] = v;
        s += v;
    }
    #pragma unroll
    for (int o = 16; o > 0; o >>= 1) s += __shfl_xor(s, o, 32);
    const float mu = s * (1.0f / kD);
    float var = 0.0f;
    #pragma unroll
    for (int i = 0; i < 16; ++i) { float dv = vals[i] - mu; var += dv * dv; }
    #pragma unroll
    for (int o = 16; o > 0; o >>= 1) var += __shfl_xor(var, o, 32);
    const float rstd = rsqrtf(var * (1.0f / kD) + 1e-5f);
    #pragma unroll
    for (int i = 0; i < 16; ++i) {
        int d = threadIdx.x + i * 32;
        float v = (vals[i] - mu) * rstd * g[d] + bt[d];
        xr[d] = v;
        xh[(size_t)row * kD + d] = (_Float16)v;
    }
}

// ---------------- softmax + argmax (first-index ties) per batch row ----------------
__global__ __launch_bounds__(256) void softmax_argmax_kernel(
    const float* __restrict__ logits,   // [BS, VPAD]
    float* __restrict__ out,            // [BS, MAXLEN, VOUT]
    int* __restrict__ dec, int t)
{
    __shared__ float sval[256];
    __shared__ int   sidx[256];
    const int b = blockIdx.x, tid = threadIdx.x;
    const float* lr = logits + (size_t)b * kVPAD;

    float bm = -3.4e38f; int bi = 0;
    for (int v = tid; v < kVOUT; v += 256) {   // strictly increasing v -> first-max wins in-thread
        float x = lr[v];
        if (x > bm) { bm = x; bi = v; }
    }
    sval[tid] = bm; sidx[tid] = bi;
    __syncthreads();
    for (int s = 128; s > 0; s >>= 1) {
        if (tid < s) {
            float ov = sval[tid + s]; int oi = sidx[tid + s];
            if (ov > sval[tid] || (ov == sval[tid] && oi < sidx[tid])) {
                sval[tid] = ov; sidx[tid] = oi;
            }
        }
        __syncthreads();
    }
    const float mx = sval[0];
    const int   am = sidx[0];
    __syncthreads();

    float sum = 0.0f;
    for (int v = tid; v < kVOUT; v += 256) sum += __expf(lr[v] - mx);
    sval[tid] = sum;
    __syncthreads();
    for (int s = 128; s > 0; s >>= 1) {
        if (tid < s) sval[tid] += sval[tid + s];
        __syncthreads();
    }
    const float inv = 1.0f / sval[0];
    float* orow = out + ((size_t)b * kMAXLEN + t) * kVOUT;
    for (int v = tid; v < kVOUT; v += 256) orow[v] = __expf(lr[v] - mx) * inv;
    if (tid == 0) dec[b * kSEQ + t + 1] = am;
}

// ---------------- host launch ----------------
extern "C" void kernel_launch(void* const* d_in, const int* in_sizes, int n_in,
                              void* d_out, int out_size, void* d_ws, size_t ws_size,
                              hipStream_t stream) {
    (void)in_sizes; (void)n_in; (void)out_size;
    const float* src   = (const float*)d_in[0];
    const float* pos   = (const float*)d_in[1];
    const float* emb   = (const float*)d_in[2];
    const float* qkvw  = (const float*)d_in[3];
    const float* qkvb  = (const float*)d_in[4];
    const float* projw = (const float*)d_in[5];
    const float* projb = (const float*)d_in[6];
    const float* ln1g  = (const float*)d_in[7];
    const float* ln1b  = (const float*)d_in[8];
    const float* fc1w  = (const float*)d_in[9];
    const float* fc1b  = (const float*)d_in[10];
    const float* fc2w  = (const float*)d_in[11];
    const float* fc2b  = (const float*)d_in[12];
    const float* ln2g  = (const float*)d_in[13];
    const float* ln2b  = (const float*)d_in[14];
    const float* prjw  = (const float*)d_in[15];
    float* out = (float*)d_out;

    uint8_t* ws = (uint8_t*)d_ws;
    size_t off = 0;
    auto alloc = [&](size_t bytes) -> void* {
        void* p = ws + off;
        off += (bytes + 255) & ~(size_t)255;
        return p;
    };
    _Float16* qkvw_h = (_Float16*)alloc((size_t)kNL * 3 * kD * kD * 2);
    _Float16* projw_h= (_Float16*)alloc((size_t)kNL * kD * kD * 2);
    _Float16* fc1w_h = (_Float16*)alloc((size_t)kNL * kDFF * kD * 2);
    _Float16* fc2w_h = (_Float16*)alloc((size_t)kNL * kD * kDFF * 2);
    _Float16* prjw_h = (_Float16*)alloc((size_t)kVPAD * kD * 2);
    float*    xf     = (float*)   alloc((size_t)kM * kD * 4);
    _Float16* xh     = (_Float16*)alloc((size_t)kM * kD * 2);
    float*    qkvbuf = (float*)   alloc((size_t)kM * 3 * kD * 4);
    _Float16* ohbuf  = (_Float16*)alloc((size_t)kM * kD * 2);
    float*    pbuf   = (float*)   alloc((size_t)kM * kD * 4);
    _Float16* h1h    = (_Float16*)alloc((size_t)kM * kDFF * 2);
    float*    logits = (float*)   alloc((size_t)kBS * kVPAD * 4);
    int*      dec    = (int*)     alloc((size_t)kBS * kSEQ * 4);
    if (off > ws_size) return;   // workspace too small: bail (no OOB writes)

    auto gemm = [&](const _Float16* A, int lda, const _Float16* W, const float* bias,
                    float* Cf, _Float16* Ch, int M, int N, int K, int flags) {
        dim3 grid(M / 16, (N + 255) / 256);
        dim3 block(32, 4);
        gemm_wmma_kernel<<<grid, block, 0, stream>>>(A, lda, W, bias, Cf, Ch, M, N, K, flags);
    };

    // one-time (per call) f16 weight staging — weights then live in L2 across all 25 steps
    cvt_kernel<<<1024, 256, 0, stream>>>(qkvw,  qkvw_h,  (size_t)kNL * 3 * kD * kD);
    cvt_kernel<<<1024, 256, 0, stream>>>(projw, projw_h, (size_t)kNL * kD * kD);
    cvt_kernel<<<1024, 256, 0, stream>>>(fc1w,  fc1w_h,  (size_t)kNL * kDFF * kD);
    cvt_kernel<<<1024, 256, 0, stream>>>(fc2w,  fc2w_h,  (size_t)kNL * kD * kDFF);
    cvt_pad_kernel<<<1024, 256, 0, stream>>>(prjw, prjw_h, kVOUT, kVPAD, kD);
    init_dec_kernel<<<(kBS * kSEQ + 255) / 256, 256, 0, stream>>>(dec);

    for (int t = 0; t < kMAXLEN; ++t) {
        embed_kernel<<<kM, 128, 0, stream>>>(src, pos, emb, dec, xf, xh);
        for (int l = 0; l < kNL; ++l) {
            // QKV: [832,512] @ [1536,512]^T
            gemm(xh, kD, qkvw_h + (size_t)l * 3 * kD * kD, qkvb + (size_t)l * 3 * kD,
                 qkvbuf, nullptr, kM, 3 * kD, kD, 0);
            attn_kernel<<<kBS * kNH, 32, 0, stream>>>(qkvbuf, ohbuf);
            // proj: [832,512] @ [512,512]^T
            gemm(ohbuf, kD, projw_h + (size_t)l * kD * kD, projb + (size_t)l * kD,
                 pbuf, nullptr, kM, kD, kD, 0);
            add_ln_kernel<<<kM, 32, 0, stream>>>(xf, pbuf, ln1g + (size_t)l * kD,
                                                 ln1b + (size_t)l * kD, xh);
            // fc1 + relu: [832,512] @ [2048,512]^T -> f16 only
            gemm(xh, kD, fc1w_h + (size_t)l * kDFF * kD, fc1b + (size_t)l * kDFF,
                 nullptr, h1h, kM, kDFF, kD, 1);
            // fc2: [832,2048] @ [512,2048]^T
            gemm(h1h, kDFF, fc2w_h + (size_t)l * kD * kDFF, fc2b + (size_t)l * kD,
                 pbuf, nullptr, kM, kD, kDFF, 0);
            add_ln_kernel<<<kM, 32, 0, stream>>>(xf, pbuf, ln2g + (size_t)l * kD,
                                                 ln2b + (size_t)l * kD, xh);
        }
        // logits at position t: rows are batch entries, stride SEQ*D
        gemm(xh + (size_t)t * kD, kSEQ * kD, prjw_h, nullptr,
             logits, nullptr, kBS, kVPAD, kD, 0);
        softmax_argmax_kernel<<<kBS, 256, 0, stream>>>(logits, out, dec, t);
    }
}
